// TimerXL_5506148073645
// MI455X (gfx1250) — matrix operand order repaired and checked by reading, not compiled, not run on hardware
//
#include <hip/hip_runtime.h>

// ---------------------------------------------------------------------------
// Types for CDNA5 WMMA (wave32, bf16 16x16x32, f32 accumulate)
// ---------------------------------------------------------------------------
typedef __attribute__((ext_vector_type(16))) __bf16        v16bf;
typedef __attribute__((ext_vector_type(8)))  float         v8f;
typedef __attribute__((ext_vector_type(4)))  unsigned int  uint4v;

typedef unsigned short bfraw;   // raw bf16 storage (avoid scalar __bf16 ops)

union FragU { uint4v u[2]; v16bf v; };

// Problem constants (match setup_inputs: B=2, L=2048, D=512, H=8, E=64,
// n_vars=32, n_tokens=64)
constexpr int Bc   = 2;
constexpr int Lc   = 2048;
constexpr int Dc   = 512;
constexpr int Hc   = 8;
constexpr int Ec   = 64;
constexpr int NVAR = 32;
constexpr int NTOK = 64;
constexpr int NROWS = Bc * Lc;          // 4096
constexpr float SCALE = 0.125f;         // 1/sqrt(E)

#define WMMA_BF16(a, b, c) \
  __builtin_amdgcn_wmma_f32_16x16x32_bf16(false, (a), false, (b), (short)0, (c), false, false)

__device__ __forceinline__ bfraw f2bf(float f) {
  unsigned int u = __float_as_uint(f);
  u += 0x7fffu + ((u >> 16) & 1u);       // round-to-nearest-even
  return (bfraw)(u >> 16);
}

// A fragment (16x32 bf16): lane holds row M = row + (lane&15).
// K layout per lane: chunks [k0+ko .. +8) and [k0+16+ko .. +8), ko = (lane>=16)*8.
__device__ __forceinline__ v16bf load_fragA(const bfraw* base, int ld, int row,
                                            int k0, int lane) {
  const bfraw* p = base + (size_t)(row + (lane & 15)) * ld + k0 + ((lane >> 4) << 3);
  FragU f;
  f.u[0] = *(const uint4v*)(p);
  f.u[1] = *(const uint4v*)(p + 16);
  return f.v;
}

// B fragment (32x16 bf16): lane holds column N = col + (lane&15).
// K layout per lane: chunks [k0+ko .. +8) and [k0+16+ko .. +8), ko = (lane>=16)*8.  Source must be
// stored so that K is contiguous per column (i.e. row-major W[n][k], or V^T).
__device__ __forceinline__ v16bf load_fragB(const bfraw* base, int ld, int col,
                                            int k0, int lane) {
  const bfraw* p = base + (size_t)(col + (lane & 15)) * ld + k0 + ((lane >> 4) << 3);
  FragU f;
  f.u[0] = *(const uint4v*)(p);
  f.u[1] = *(const uint4v*)(p + 16);
  return f.v;
}

// ---------------------------------------------------------------------------
// fp32 -> bf16 conversion
// ---------------------------------------------------------------------------
__global__ __launch_bounds__(256) void cvt_bf16_kernel(const float* __restrict__ src,
                                                       bfraw* __restrict__ dst, int n) {
  int i = blockIdx.x * blockDim.x + threadIdx.x;
  if (i < n) dst[i] = f2bf(src[i]);
}

// ---------------------------------------------------------------------------
// WMMA GEMM:  Y[m][n] = sum_k X[m][k] * W[n][k] + bias[n]
//   mode 0: Q -> RoPE, *SCALE, store [B,H,L,E] bf16   (outHead)
//   mode 1: K -> RoPE,          store [B,H,L,E] bf16   (outHead)
//   mode 2: V ->                store [B,H,E,L] bf16   (outVt, transposed)
//   mode 3: O ->                store fp32 [m][n]      (outF = d_out)
// Block: 256 threads = 8 waves (2 row x 4 col), wave tile 32x32 (2x2 WMMA
// register blocking -> 4 WMMA per 4 fragment loads), block tile 64x128.
// ---------------------------------------------------------------------------
__global__ __launch_bounds__(256) void gemm_wmma_kernel(
    const bfraw* __restrict__ X, const bfraw* __restrict__ W,
    const float* __restrict__ bias, int mode,
    bfraw* __restrict__ outHead, bfraw* __restrict__ outVt,
    float* __restrict__ outF) {
  const int lane = threadIdx.x & 31;
  const int wave = threadIdx.x >> 5;
  const int row0 = blockIdx.y * 64  + (wave >> 2) * 32;   // 2 wave-rows
  const int col0 = blockIdx.x * 128 + (wave & 3) * 32;    // 4 wave-cols

  v8f acc00 = {0.f, 0.f, 0.f, 0.f, 0.f, 0.f, 0.f, 0.f};
  v8f acc01 = acc00, acc10 = acc00, acc11 = acc00;

  for (int k0 = 0; k0 < Dc; k0 += 32) {
    v16bf b0 = load_fragB(W, Dc, col0,      k0, lane);
    v16bf b1 = load_fragB(W, Dc, col0 + 16, k0, lane);
    v16bf a0 = load_fragA(X, Dc, row0,      k0, lane);
    v16bf a1 = load_fragA(X, Dc, row0 + 16, k0, lane);
    acc00 = WMMA_BF16(a0, b0, acc00);
    acc01 = WMMA_BF16(a0, b1, acc01);
    acc10 = WMMA_BF16(a1, b0, acc10);
    acc11 = WMMA_BF16(a1, b1, acc11);
  }

  const int ho = (lane >> 4) << 3;        // row offset 0 or 8 within tile

#pragma unroll
  for (int tr = 0; tr < 2; ++tr) {
#pragma unroll
    for (int tc = 0; tc < 2; ++tc) {
      v8f acc = tr ? (tc ? acc11 : acc10) : (tc ? acc01 : acc00);
      const int n = col0 + tc * 16 + (lane & 15);
      const float bval = bias[n];
#pragma unroll
      for (int r = 0; r < 8; ++r) {
        int   m   = row0 + tr * 16 + r + ho;
        float val = acc[r] + bval;
        if (mode <= 1) {                   // Q or K: interleaved RoPE on e<32
          int bb = m / Lc, l = m % Lc;
          int h = n >> 6, e = n & 63;
          float partner = __shfl_xor(val, 1, 32);  // pre-rope value of pair
          if (e < 32) {
            float theta = __powf(10000.0f, -(float)(e & ~1) * (1.0f / 32.0f));
            float ang   = (float)l * theta;
            float c = __cosf(ang), s = __sinf(ang);
            float rot = (e & 1) ? partner : -partner;
            val = c * val + s * rot;
          }
          if (mode == 0) val *= SCALE;     // fold 1/sqrt(E) into Q
          outHead[(((size_t)bb * Hc + h) * Lc + l) * Ec + e] = f2bf(val);
        } else if (mode == 2) {            // V transposed: [B,H,E,L]
          int bb = m / Lc, l = m % Lc;
          int h = n >> 6, e = n & 63;
          outVt[(((size_t)bb * Hc + h) * Ec + e) * Lc + l] = f2bf(val);
        } else {                           // final output projection (fp32)
          outF[(size_t)m * Dc + n] = val;
        }
      }
    }
  }
}

// ---------------------------------------------------------------------------
// Attention: one workgroup per (b, h, var). 8 waves, each owning 2 of the
// 16 (16x16) tiles of the 64x64 working set (tn = wave&3, tm = (wave>>2)+2*half).
// ---------------------------------------------------------------------------
__device__ __forceinline__ void compute_S_tiles(const bfraw* qh, const bfraw* kh,
                                                int qrow0, int krow0,
                                                int wave, int lane, float* Ssh) {
  const int tn  = wave & 3;
  const int tmA = wave >> 2;
#pragma unroll
  for (int half = 0; half < 2; ++half) {
    int tm = tmA + half * 2;
    v8f acc = {0.f, 0.f, 0.f, 0.f, 0.f, 0.f, 0.f, 0.f};
#pragma unroll
    for (int e0 = 0; e0 < Ec; e0 += 32) {
      v16bf a  = load_fragA(qh, Ec, qrow0 + tm * 16, e0, lane);
      v16bf bb = load_fragB(kh, Ec, krow0 + tn * 16, e0, lane);
      acc = WMMA_BF16(a, bb, acc);
    }
    int ho = (lane >> 4) << 3;
    int cc = tn * 16 + (lane & 15);
#pragma unroll
    for (int r = 0; r < 8; ++r)
      Ssh[(tm * 16 + r + ho) * 64 + cc] = acc[r];
  }
}

__global__ __launch_bounds__(256) void attn_wmma_kernel(
    const bfraw* __restrict__ Qs, const bfraw* __restrict__ Ks,
    const bfraw* __restrict__ Vt, const float* __restrict__ bias_emb,
    bfraw* __restrict__ AttnOut) {
  __shared__ float Ssh[64 * 64];     // score tile (C-layout dump)
  __shared__ bfraw Psh[64 * 64];     // probs, row-major bf16 (A-frag loadable)
  __shared__ float mrow[64], lrow[64];

  const int lane = threadIdx.x & 31;
  const int wave = threadIdx.x >> 5;
  const int v = blockIdx.x % NVAR;
  const int h = (blockIdx.x / NVAR) % Hc;
  const int b = blockIdx.x / (NVAR * Hc);

  const bfraw* qh = Qs + ((size_t)b * Hc + h) * Lc * Ec;
  const bfraw* kh = Ks + ((size_t)b * Hc + h) * Lc * Ec;
  const bfraw* vh = Vt + ((size_t)b * Hc + h) * (size_t)Ec * Lc;

  const float bias0 = SCALE * bias_emb[0 * Hc + h];   // different var
  const float bias1 = SCALE * bias_emb[1 * Hc + h];   // same var
  const int  qrow0   = v * NTOK;
  const bool lastVar = (v == NVAR - 1);
  const int  nkb     = lastVar ? NVAR : 1;

  // softmax cooperative mapping: 4 threads per row, 16 cols each
  const int rr  = threadIdx.x >> 2;       // row 0..63
  const int sub = threadIdx.x & 3;        // column quarter

  if (threadIdx.x < 64) { mrow[threadIdx.x] = -1e30f; lrow[threadIdx.x] = 0.f; }
  __syncthreads();

  // ---- pass 1: row max / denominator over all allowed key blocks ----
  for (int i = 0; i < nkb; ++i) {
    int kb = lastVar ? i : v;
    compute_S_tiles(qh, kh, qrow0, kb * NTOK, wave, lane, Ssh);
    __syncthreads();
    {
      bool same = (kb == v);
      float bv  = same ? bias1 : bias0;
      int cmax  = same ? (rr + 1) : 64;   // causal inside own block
      float bm = -1e30f;
#pragma unroll
      for (int j = 0; j < 16; ++j) {
        int c = sub * 16 + j;
        if (c < cmax) bm = fmaxf(bm, Ssh[rr * 64 + c] + bv);
      }
      bm = fmaxf(bm, __shfl_xor(bm, 1, 32));
      bm = fmaxf(bm, __shfl_xor(bm, 2, 32));
      float m_old = mrow[rr];
      float m_new = fmaxf(m_old, bm);
      float ls = 0.f;
#pragma unroll
      for (int j = 0; j < 16; ++j) {
        int c = sub * 16 + j;
        if (c < cmax) ls += __expf(Ssh[rr * 64 + c] + bv - m_new);
      }
      ls += __shfl_xor(ls, 1, 32);
      ls += __shfl_xor(ls, 2, 32);
      if (sub == 0) {
        lrow[rr] = lrow[rr] * __expf(m_old - m_new) + ls;
        mrow[rr] = m_new;
      }
    }
    __syncthreads();
  }

  // ---- pass 2: O += exp(S - m) @ V, normalize by l at the end ----
  v8f accO0 = {0.f, 0.f, 0.f, 0.f, 0.f, 0.f, 0.f, 0.f};
  v8f accO1 = accO0;
  const int tn  = wave & 3;
  const int tmA = wave >> 2;

  for (int i = 0; i < nkb; ++i) {
    int kb = lastVar ? i : v;
    compute_S_tiles(qh, kh, qrow0, kb * NTOK, wave, lane, Ssh);
    __syncthreads();
    {
      bool same = (kb == v);
      float bv  = same ? bias1 : bias0;
      int cmax  = same ? (rr + 1) : 64;
      float mr  = mrow[rr];
#pragma unroll
      for (int j = 0; j < 16; ++j) {
        int c = sub * 16 + j;
        float p = (c < cmax) ? __expf(Ssh[rr * 64 + c] + bv - mr) : 0.f;
        Psh[rr * 64 + c] = f2bf(p);
      }
    }
    __syncthreads();
#pragma unroll
    for (int half = 0; half < 2; ++half) {
      int tm = tmA + half * 2;
      v8f acc = half ? accO1 : accO0;
#pragma unroll
      for (int s0 = 0; s0 < NTOK; s0 += 32) {
        v16bf a  = load_fragA(Psh, 64, tm * 16, s0, lane);            // LDS
        v16bf bb = load_fragB(vh, Lc, tn * 16, kb * NTOK + s0, lane); // V^T rows
        acc = WMMA_BF16(a, bb, acc);
      }
      if (half) accO1 = acc; else accO0 = acc;
    }
    __syncthreads();
  }

  // ---- epilogue: normalize and store [B, L, H*E] bf16 ----
  const int ho = (lane >> 4) << 3;
  const int ee = tn * 16 + (lane & 15);
#pragma unroll
  for (int half = 0; half < 2; ++half) {
    int tm = tmA + half * 2;
    v8f acc = half ? accO1 : accO0;
#pragma unroll
    for (int r = 0; r < 8; ++r) {
      int m = tm * 16 + r + ho;
      float val = acc[r] / lrow[m];
      int l = qrow0 + m;
      AttnOut[((size_t)(b * Lc + l)) * Dc + h * Ec + ee] = f2bf(val);
    }
  }
}

// ---------------------------------------------------------------------------
// Host launcher
// ---------------------------------------------------------------------------
extern "C" void kernel_launch(void* const* d_in, const int* in_sizes, int n_in,
                              void* d_out, int out_size, void* d_ws, size_t ws_size,
                              hipStream_t stream) {
  (void)in_sizes; (void)n_in; (void)out_size; (void)ws_size;
  const float* x        = (const float*)d_in[0];
  const float* Wq       = (const float*)d_in[1];
  const float* bq       = (const float*)d_in[2];
  const float* Wk       = (const float*)d_in[3];
  const float* bk       = (const float*)d_in[4];
  const float* Wv       = (const float*)d_in[5];
  const float* bv       = (const float*)d_in[6];
  const float* Wo       = (const float*)d_in[7];
  const float* bo       = (const float*)d_in[8];
  const float* bias_emb = (const float*)d_in[9];
  float* out = (float*)d_out;

  // workspace layout (bf16 elements)
  const size_t XN = (size_t)NROWS * Dc;          // 4096*512
  const size_t WN = (size_t)Dc * Dc;             // 512*512
  bfraw* xbf = (bfraw*)d_ws;
  bfraw* wqb = xbf + XN;
  bfraw* wkb = wqb + WN;
  bfraw* wvb = wkb + WN;
  bfraw* wob = wvb + WN;
  bfraw* Qsb = wob + WN;                         // [B,H,L,E]
  bfraw* Ksb = Qsb + XN;                         // [B,H,L,E]
  bfraw* Vtb = Ksb + XN;                         // [B,H,E,L]
  bfraw* Att = Vtb + XN;                         // [B,L,H*E]

  cvt_bf16_kernel<<<(int)((XN + 255) / 256), 256, 0, stream>>>(x,  xbf, (int)XN);
  cvt_bf16_kernel<<<(int)((WN + 255) / 256), 256, 0, stream>>>(Wq, wqb, (int)WN);
  cvt_bf16_kernel<<<(int)((WN + 255) / 256), 256, 0, stream>>>(Wk, wkb, (int)WN);
  cvt_bf16_kernel<<<(int)((WN + 255) / 256), 256, 0, stream>>>(Wv, wvb, (int)WN);
  cvt_bf16_kernel<<<(int)((WN + 255) / 256), 256, 0, stream>>>(Wo, wob, (int)WN);

  dim3 gg(Dc / 128, NROWS / 64);
  gemm_wmma_kernel<<<gg, 256, 0, stream>>>(xbf, wqb, bq, 0, Qsb, nullptr, nullptr);
  gemm_wmma_kernel<<<gg, 256, 0, stream>>>(xbf, wkb, bk, 1, Ksb, nullptr, nullptr);
  gemm_wmma_kernel<<<gg, 256, 0, stream>>>(xbf, wvb, bv, 2, nullptr, Vtb, nullptr);

  attn_wmma_kernel<<<Bc * Hc * NVAR, 256, 0, stream>>>(Qsb, Ksb, Vtb, bias_emb, Att);

  gemm_wmma_kernel<<<gg, 256, 0, stream>>>(Att, wob, bo, 3, nullptr, nullptr, out);
}
